// GATv2Net_67216238182419
// MI455X (gfx1250) — compile-verified, matrix operand order checked by reading, not run on hardware
//
#include <hip/hip_runtime.h>
#include <math.h>

#define NNODES  50000
#define NEDGES  800000
#define INDIM   1280
#define HIDDIM  256
#define NHEAD   4
#define ETOT    (NEDGES + NNODES)   // self-loops appended
#define NEG_SLOPE 0.2f

typedef __bf16 bf16_t;
typedef __bf16 v8bf   __attribute__((ext_vector_type(8)));
typedef __bf16 v16bf  __attribute__((ext_vector_type(16)));
typedef float  v8f    __attribute__((ext_vector_type(8)));
typedef int    v2i    __attribute__((ext_vector_type(2)));

// ---- CDNA5 async global->LDS path (guarded; sync fallback otherwise)
#if defined(__has_builtin)
#  if __has_builtin(__builtin_amdgcn_global_load_async_to_lds_b64) && \
      __has_builtin(__builtin_amdgcn_s_wait_asynccnt)
#    define USE_ASYNC 1
#  endif
#endif
#ifndef USE_ASYNC
#  define USE_ASYNC 0
#endif

__device__ __forceinline__ void async_cp_b64(const bf16_t* g, bf16_t* l) {
#if USE_ASYNC
  // prototype (from clang diagnostic): (v2i addrspace(1)*, v2i addrspace(3)*, imm, imm)
  __builtin_amdgcn_global_load_async_to_lds_b64(
      (__attribute__((address_space(1))) v2i*)g,
      (__attribute__((address_space(3))) v2i*)l, 0, 0);
#else
  *(uint2*)l = *(const uint2*)g;
#endif
}
__device__ __forceinline__ void wait_async_le1() {
#if USE_ASYNC
  __builtin_amdgcn_s_wait_asynccnt(1);
#endif
}
__device__ __forceinline__ void wait_async_le0() {
#if USE_ASYNC
  __builtin_amdgcn_s_wait_asynccnt(0);
#endif
}

__device__ __forceinline__ bf16_t f2bf(float f) {
  unsigned int u = __float_as_uint(f);
  u += 0x7FFFu + ((u >> 16) & 1u);          // round-to-nearest-even
  unsigned short s = (unsigned short)(u >> 16);
  return __builtin_bit_cast(bf16_t, s);
}

__device__ __forceinline__ float lrelu(float x) { return x > 0.f ? x : NEG_SLOPE * x; }

__device__ __forceinline__ void atomicMaxF(float* addr, float val) {
  unsigned int* ua = (unsigned int*)addr;
  unsigned int old = *ua;
  while (__uint_as_float(old) < val) {
    unsigned int assumed = old;
    old = atomicCAS(ua, assumed, __float_as_uint(val));
    if (old == assumed) break;
  }
}

// ---------------------------------------------------------------- conversions
__global__ __launch_bounds__(256) void k_f32_to_bf16(
    const float* __restrict__ in, bf16_t* __restrict__ out, long long n) {
  long long i = (long long)blockIdx.x * 256 + threadIdx.x;
  if (i < n) out[i] = f2bf(in[i]);
}

// Pack W[K,N] (row-major f32) into bf16 panels [K/32][N][32] so each lane of a
// WMMA wave reads its 16 K-values with two 16B vector loads.
__global__ __launch_bounds__(256) void k_pack_w(
    const float* __restrict__ W, bf16_t* __restrict__ Wp, int K, int N) {
  int i = blockIdx.x * 256 + threadIdx.x;
  if (i >= K * N) return;
  int k = i / N, n = i - k * N;
  Wp[(((size_t)(k >> 5) * N + n) << 5) + (k & 31)] = f2bf(W[i]);
}

__global__ __launch_bounds__(256) void k_fill(float* __restrict__ p, float v, int n) {
  int i = blockIdx.x * 256 + threadIdx.x;
  if (i < n) p[i] = v;
}

// ---------------------------------------------------------------- WMMA GEMM
// C[M,N] = A[M,K](bf16,row-major) @ B(packed bf16 panels) + bias[N]  (f32 acc)
// Block = 256 threads = 8 waves (2 M-warps x 4 N-warps), block tile 32 x 256.
// A panels (32 rows x 32 k, 2KB) are double-buffered in LDS via async
// global->LDS copies (ASYNCcnt) so all 4 N-warps share one A fetch.
__global__ __launch_bounds__(256) void k_wmma_gemm_bias(
    const bf16_t* __restrict__ A, const bf16_t* __restrict__ Bp,
    const float* __restrict__ bias, float* __restrict__ C,
    int M, int K, int N)
{
  __shared__ __align__(16) bf16_t Apan[2][32 * 32];   // 2 x 2KB

  const int tid   = threadIdx.x;
  const int lane  = tid & 31;
  const int wave  = tid >> 5;
  const int mblk  = blockIdx.x * 32;
  const int mwarp = wave >> 2;               // 0..1
  const int nwarp = wave & 3;                // 0..3
  const int m0    = mblk + mwarp * 16;
  const int n0    = nwarp * 64;
  const bool mvalid = (m0 < M);              // ragged last block: no early return

  const int lm    = lane & 15;
  const int khalf = (lane >> 4) << 3;        // 0 or 8 (16-bit A/B lane layout)

  // copy mapping: thread t stages row t/8, 4 bf16 (8B) at col (t%8)*4
  const int  crow   = tid >> 3;              // 0..31
  const int  ccol   = (tid & 7) << 2;        // 0..28 step 4 (bf16 elems)
  const int  grow   = mblk + crow;
  const bool cvalid = (grow < M);
  const bf16_t* gsrc = A + (size_t)grow * K + ccol;
  bf16_t* l0 = &Apan[0][crow * 32 + ccol];
  bf16_t* l1 = &Apan[1][crow * 32 + ccol];

  v8f acc[4] = {};
  const int nch = K >> 5;                    // k-chunks of 32

  auto do_chunk = [&](int i) {
    const bf16_t* ap = &Apan[i & 1][(mwarp * 16 + lm) * 32 + khalf];
    v8bf alo = *(const v8bf*)ap;
    v8bf ahi = *(const v8bf*)(ap + 16);
    v16bf a = __builtin_shufflevector(alo, ahi,
                0,1,2,3,4,5,6,7,8,9,10,11,12,13,14,15);
    const bf16_t* bbase = Bp + (((size_t)i * N + n0 + lm) << 5) + khalf;
    #pragma unroll
    for (int j = 0; j < 4; ++j) {
      const bf16_t* bp = bbase + ((size_t)j << 9);   // +16 columns * 32
      v8bf blo = *(const v8bf*)bp;
      v8bf bhi = *(const v8bf*)(bp + 16);
      v16bf b = __builtin_shufflevector(blo, bhi,
                  0,1,2,3,4,5,6,7,8,9,10,11,12,13,14,15);
      acc[j] = __builtin_amdgcn_wmma_f32_16x16x32_bf16(
          false, a, false, b, (short)0, acc[j], false, false);
    }
  };

  if (cvalid) async_cp_b64(gsrc, l0);                // prologue: chunk 0
  for (int i = 0; i < nch - 1; ++i) {
    if (cvalid) async_cp_b64(gsrc + ((i + 1) << 5), (i & 1) ? l0 : l1);
    wait_async_le1();                                // chunk i landed (in-order)
    __syncthreads();
    if (mvalid) do_chunk(i);
    __syncthreads();                                 // done reading buf[i&1]
  }
  wait_async_le0();                                  // last chunk
  __syncthreads();
  if (mvalid) do_chunk(nch - 1);

  if (!mvalid) return;
  const int rbase = m0 + ((lane >> 4) << 3);         // lanes 16-31 -> M+8
  #pragma unroll
  for (int j = 0; j < 4; ++j) {
    const int n = n0 + (j << 4) + lm;
    const float bv = bias[n];
    #pragma unroll
    for (int v = 0; v < 8; ++v)
      C[(size_t)(rbase + v) * N + n] = acc[j][v] + bv;
  }
}

// ---------------------------------------------------------------- edge stage
// One wave per edge; lanes 8h..8h+7 cover head h (8 floats each, float4 x2).
__global__ __launch_bounds__(256) void k_edge_logits(
    const float* __restrict__ xl, const float* __restrict__ xr,
    const int* __restrict__ esrc, const int* __restrict__ edst,
    const float* __restrict__ att, float* __restrict__ logits,
    float* __restrict__ mmax)
{
  const int e = blockIdx.x * 8 + (threadIdx.x >> 5);
  if (e >= ETOT) return;
  const int lane = threadIdx.x & 31;
  const int src = (e < NEDGES) ? esrc[e] : (e - NEDGES);
  const int dst = (e < NEDGES) ? edst[e] : (e - NEDGES);
  const int base = lane << 3;
  const float4* pl = (const float4*)(xl + (size_t)src * HIDDIM + base);
  const float4* pr = (const float4*)(xr + (size_t)dst * HIDDIM + base);
  const float4* pa = (const float4*)(att + base);
  float s = 0.f;
  #pragma unroll
  for (int q = 0; q < 2; ++q) {
    float4 a = pl[q], b = pr[q], w = pa[q];
    s += lrelu(a.x + b.x) * w.x + lrelu(a.y + b.y) * w.y
       + lrelu(a.z + b.z) * w.z + lrelu(a.w + b.w) * w.w;
  }
  s += __shfl_xor(s, 1, 32);
  s += __shfl_xor(s, 2, 32);
  s += __shfl_xor(s, 4, 32);
  if ((lane & 7) == 0) {
    const int h = lane >> 3;
    logits[(size_t)e * NHEAD + h] = s;
    atomicMaxF(mmax + (size_t)dst * NHEAD + h, s);
  }
}

__global__ __launch_bounds__(256) void k_edge_exp(
    const float* __restrict__ logits, const int* __restrict__ edst,
    const float* __restrict__ mmax, float* __restrict__ alpha,
    float* __restrict__ denom)
{
  int idx = blockIdx.x * 256 + threadIdx.x;
  if (idx >= ETOT * NHEAD) return;
  int e = idx >> 2, h = idx & 3;
  int dst = (e < NEDGES) ? edst[e] : (e - NEDGES);
  float a = expf(logits[idx] - mmax[(size_t)dst * NHEAD + h]);
  alpha[idx] = a;                       // unnormalized, fixed up next kernel
  atomicAdd(denom + (size_t)dst * NHEAD + h, a);
}

__global__ __launch_bounds__(256) void k_edge_aggregate(
    const float* __restrict__ xl,
    const int* __restrict__ esrc, const int* __restrict__ edst,
    const float* __restrict__ denom, float* __restrict__ alpha,
    float* __restrict__ acc)
{
  const int e = blockIdx.x * 8 + (threadIdx.x >> 5);
  if (e >= ETOT) return;
  const int lane = threadIdx.x & 31;
  const int src = (e < NEDGES) ? esrc[e] : (e - NEDGES);
  const int dst = (e < NEDGES) ? edst[e] : (e - NEDGES);
  const int h = lane >> 3;
  const float a = alpha[(size_t)e * NHEAD + h];
  const float d = denom[(size_t)dst * NHEAD + h];
  const float al = a / (d + 1e-16f);
  if ((lane & 7) == 0) alpha[(size_t)e * NHEAD + h] = al;   // normalized out
  const int base = lane << 3;
  const float* ps = xl  + (size_t)src * HIDDIM + base;
  float*       pd = acc + (size_t)dst * HIDDIM + base;
  #pragma unroll
  for (int q = 0; q < 8; ++q) atomicAdd(pd + q, ps[q] * al);
}

__global__ __launch_bounds__(256) void k_finalize(
    const float* __restrict__ acc, const float* __restrict__ bias,
    const float* __restrict__ hin, float* __restrict__ hout)
{
  int i = blockIdx.x * 256 + threadIdx.x;
  if (i >= NNODES * HIDDIM) return;
  float v = acc[i] + bias[i & (HIDDIM - 1)] + hin[i];
  hout[i] = v > 0.f ? v : (expf(v) - 1.f);   // ELU
}

__global__ __launch_bounds__(256) void k_classifier(
    const float* __restrict__ h, const float* __restrict__ W,
    const float* __restrict__ b, float* __restrict__ preds)
{
  const int n = blockIdx.x * 8 + (threadIdx.x >> 5);
  if (n >= NNODES) return;
  const int lane = threadIdx.x & 31;
  const float4* ph = (const float4*)(h + (size_t)n * HIDDIM + (lane << 3));
  const float4* pw = (const float4*)(W + (lane << 3));
  float s = 0.f;
  #pragma unroll
  for (int q = 0; q < 2; ++q) {
    float4 a = ph[q], w = pw[q];
    s += a.x * w.x + a.y * w.y + a.z * w.z + a.w * w.w;
  }
  #pragma unroll
  for (int off = 16; off > 0; off >>= 1) s += __shfl_xor(s, off, 32);
  if (lane == 0) preds[n] = s + b[0];
}

// ---------------------------------------------------------------- launcher
extern "C" void kernel_launch(void* const* d_in, const int* in_sizes, int n_in,
                              void* d_out, int out_size, void* d_ws, size_t ws_size,
                              hipStream_t stream)
{
  (void)in_sizes; (void)n_in; (void)out_size; (void)ws_size;

  const float* x     = (const float*)d_in[0];
  const int*   eidx  = (const int*)  d_in[1];
  const float* enc_W = (const float*)d_in[2];
  const float* enc_b = (const float*)d_in[3];
  const float* Wl0   = (const float*)d_in[4];
  const float* bl0   = (const float*)d_in[5];
  const float* Wr0   = (const float*)d_in[6];
  const float* br0   = (const float*)d_in[7];
  const float* att0  = (const float*)d_in[8];
  const float* bias0 = (const float*)d_in[9];
  const float* Wl1   = (const float*)d_in[10];
  const float* bl1   = (const float*)d_in[11];
  const float* Wr1   = (const float*)d_in[12];
  const float* br1   = (const float*)d_in[13];
  const float* att1  = (const float*)d_in[14];
  const float* bias1 = (const float*)d_in[15];
  const float* clfW  = (const float*)d_in[16];
  const float* clfb  = (const float*)d_in[17];

  const int* esrc = eidx;            // edge_index[0]
  const int* edst = eidx + NEDGES;   // edge_index[1]

  // ---- workspace carving
  char* ws = (char*)d_ws;
  size_t off = 0;
  auto take = [&](size_t bytes) -> char* {
    char* p = ws + off;
    off += (bytes + 255) & ~(size_t)255;
    return p;
  };
  float*  hA     = (float*) take(sizeof(float)  * (size_t)NNODES * HIDDIM);
  float*  hB     = (float*) take(sizeof(float)  * (size_t)NNODES * HIDDIM);
  float*  xl     = (float*) take(sizeof(float)  * (size_t)NNODES * HIDDIM);
  float*  xr     = (float*) take(sizeof(float)  * (size_t)NNODES * HIDDIM);
  float*  accb   = (float*) take(sizeof(float)  * (size_t)NNODES * HIDDIM);
  float*  mmax   = (float*) take(sizeof(float)  * (size_t)NNODES * NHEAD);
  float*  denom  = (float*) take(sizeof(float)  * (size_t)NNODES * NHEAD);
  float*  logits = (float*) take(sizeof(float)  * (size_t)ETOT   * NHEAD);
  bf16_t* x_bf   = (bf16_t*)take(sizeof(bf16_t) * (size_t)NNODES * INDIM);
  bf16_t* h_bf   = (bf16_t*)take(sizeof(bf16_t) * (size_t)NNODES * HIDDIM);
  bf16_t* encWp  = (bf16_t*)take(sizeof(bf16_t) * (size_t)INDIM  * HIDDIM);
  bf16_t* Wl0p   = (bf16_t*)take(sizeof(bf16_t) * (size_t)HIDDIM * HIDDIM);
  bf16_t* Wr0p   = (bf16_t*)take(sizeof(bf16_t) * (size_t)HIDDIM * HIDDIM);
  bf16_t* Wl1p   = (bf16_t*)take(sizeof(bf16_t) * (size_t)HIDDIM * HIDDIM);
  bf16_t* Wr1p   = (bf16_t*)take(sizeof(bf16_t) * (size_t)HIDDIM * HIDDIM);

  const dim3 B256(256);
  const int gemm_grid = (NNODES + 31) / 32;                 // 1563 blocks
  const int nh_grid   = (NNODES * NHEAD + 255) / 256;
  const int nc_grid   = (NNODES * HIDDIM + 255) / 256;
  const int edge_grid = (ETOT + 7) / 8;
  const int eh_grid   = (ETOT * NHEAD + 255) / 256;

  // ---- one-time precision conversion / weight packing
  k_f32_to_bf16<<<(int)(((long long)NNODES * INDIM + 255) / 256), B256, 0, stream>>>(
      x, x_bf, (long long)NNODES * INDIM);
  k_pack_w<<<(INDIM * HIDDIM + 255) / 256, B256, 0, stream>>>(enc_W, encWp, INDIM, HIDDIM);
  k_pack_w<<<(HIDDIM * HIDDIM + 255) / 256, B256, 0, stream>>>(Wl0, Wl0p, HIDDIM, HIDDIM);
  k_pack_w<<<(HIDDIM * HIDDIM + 255) / 256, B256, 0, stream>>>(Wr0, Wr0p, HIDDIM, HIDDIM);
  k_pack_w<<<(HIDDIM * HIDDIM + 255) / 256, B256, 0, stream>>>(Wl1, Wl1p, HIDDIM, HIDDIM);
  k_pack_w<<<(HIDDIM * HIDDIM + 255) / 256, B256, 0, stream>>>(Wr1, Wr1p, HIDDIM, HIDDIM);

  // ---- encoder GEMM
  k_wmma_gemm_bias<<<gemm_grid, B256, 0, stream>>>(x_bf, encWp, enc_b, hA,
                                                   NNODES, INDIM, HIDDIM);

  float* preds  = (float*)d_out;
  float* alpha0 = preds + NNODES;
  float* alpha1 = alpha0 + (size_t)ETOT * NHEAD;

  auto run_layer = [&](const float* hin, float* hout,
                       const bf16_t* Wlp, const float* bl,
                       const bf16_t* Wrp, const float* br,
                       const float* att, const float* bias,
                       float* alpha) {
    k_f32_to_bf16<<<nc_grid, B256, 0, stream>>>(hin, h_bf, (long long)NNODES * HIDDIM);
    k_wmma_gemm_bias<<<gemm_grid, B256, 0, stream>>>(h_bf, Wlp, bl, xl,
                                                     NNODES, HIDDIM, HIDDIM);
    k_wmma_gemm_bias<<<gemm_grid, B256, 0, stream>>>(h_bf, Wrp, br, xr,
                                                     NNODES, HIDDIM, HIDDIM);
    k_fill<<<nh_grid, B256, 0, stream>>>(mmax, -3.402823466e38f, NNODES * NHEAD);
    k_fill<<<nh_grid, B256, 0, stream>>>(denom, 0.f, NNODES * NHEAD);
    k_fill<<<nc_grid, B256, 0, stream>>>(accb, 0.f, NNODES * HIDDIM);
    k_edge_logits<<<edge_grid, B256, 0, stream>>>(xl, xr, esrc, edst, att, logits, mmax);
    k_edge_exp<<<eh_grid, B256, 0, stream>>>(logits, edst, mmax, alpha, denom);
    k_edge_aggregate<<<edge_grid, B256, 0, stream>>>(xl, esrc, edst, denom, alpha, accb);
    k_finalize<<<nc_grid, B256, 0, stream>>>(accb, bias, hin, hout);
  };

  run_layer(hA, hB, Wl0p, bl0, Wr0p, br0, att0, bias0, alpha0);
  run_layer(hB, hA, Wl1p, bl1, Wr1p, br1, att1, bias1, alpha1);

  k_classifier<<<(NNODES + 7) / 8, B256, 0, stream>>>(hA, clfW, clfb, preds);
}
